// Attention_21285857919576
// MI455X (gfx1250) — matrix operand order repaired and checked by reading, not compiled, not run on hardware
//
#include <hip/hip_runtime.h>
#include <hip/hip_bf16.h>

// ---------------------------------------------------------------------------
// MI455X (gfx1250) attention block: QKV proj -> RMSNorm+RoPE -> causal flash
// attention -> output proj.  All matmuls on v_wmma_f32_16x16x32_bf16 (f32
// accumulate).  Inputs pre-cast to bf16 once; GEMM tiles and the attention
// V-blocks are staged into LDS with global_load_async_to_lds_b128 (ASYNCcnt).
// ---------------------------------------------------------------------------

typedef __attribute__((ext_vector_type(16))) __bf16 v16bf;
typedef __attribute__((ext_vector_type(8)))  float  v8f;
typedef __attribute__((ext_vector_type(4)))  unsigned short us4;

union BF16Frag {            // A or B operand fragment (16 bf16 per lane)
    v16bf v;
    unsigned short u[16];
};
union F32Frag {             // C/D fragment (8 f32 per lane)
    v8f   v;
    float e[8];
};

__device__ __forceinline__ unsigned short f2bf(float f) {
    // native f32 -> bf16 conversion (hardware cvt on gfx1250)
    return __builtin_bit_cast(unsigned short, (__bf16)f);
}

__device__ __forceinline__ v8f wmma_bf16(const BF16Frag& a, const BF16Frag& b, v8f c) {
    return __builtin_amdgcn_wmma_f32_16x16x32_bf16(
        /*neg_a=*/false, a.v, /*neg_b=*/false, b.v,
        /*c_mod=*/(short)0, c, /*reuse_a=*/false, /*reuse_b=*/false);
}

__device__ __forceinline__ unsigned int lds_off(const void* p) {
    // low 32 bits of a generic pointer into the LDS aperture = LDS byte offset
    return (unsigned int)(unsigned long long)p;
}

__device__ __forceinline__ void async_load_b128(unsigned int lds_byte_off,
                                                const void* gaddr) {
    asm volatile("global_load_async_to_lds_b128 %0, %1, off"
                 :: "v"(lds_byte_off), "v"(gaddr) : "memory");
}

__device__ __forceinline__ void wait_async0() {
    asm volatile("s_wait_asynccnt 0x0" ::: "memory");
}

// ---------------------------------------------------------------------------
// Elementwise fp32 -> bf16 cast (float4 per thread), bandwidth-bound one-shot.
// ---------------------------------------------------------------------------
__global__ __launch_bounds__(256)
void cast_f32_bf16(const float* __restrict__ in, unsigned short* __restrict__ out) {
    const size_t i = (size_t)blockIdx.x * 256 + threadIdx.x;
    const float4 x = ((const float4*)in)[i];
    ((us4*)out)[i] = (us4){f2bf(x.x), f2bf(x.y), f2bf(x.z), f2bf(x.w)};
}

// ---------------------------------------------------------------------------
// C[M,N] = A[M,K] @ W[N,K]^T   (bf16 in, fp32 out).
// 256 threads (8 waves), 64x64 block tile, K-step 32.
// Tiles filled via async memory->LDS DMA; inner loop is ds_load + wmma only.
// ---------------------------------------------------------------------------
__global__ __launch_bounds__(256)
void gemm_bf16_wmma(const unsigned short* __restrict__ A,
                    const unsigned short* __restrict__ W,
                    float* __restrict__ C, int M, int N, int K) {
    __shared__ unsigned short As[64][32];
    __shared__ unsigned short Bs[64][32];

    const int tid  = threadIdx.x;
    const int lane = tid & 31;
    const int wave = tid >> 5;      // 0..7
    const int wr   = wave >> 1;     // 0..3
    const int wc   = wave & 1;      // 0..1
    const int half = lane >> 4;
    const int l16  = lane & 15;

    const size_t rowBase = (size_t)blockIdx.x * 64;
    const size_t colBase = (size_t)blockIdx.y * 64;

    F32Frag c0, c1;
    #pragma unroll
    for (int i = 0; i < 8; ++i) { c0.e[i] = 0.f; c1.e[i] = 0.f; }

    // async tile fill: each thread moves one 16B chunk of A and one of B
    const int r = tid >> 2;         // 0..63
    const int cc = (tid & 3) * 8;   // 0,8,16,24 (bf16 elements)
    const unsigned int ldsA = lds_off(&As[r][cc]);
    const unsigned int ldsB = lds_off(&Bs[r][cc]);

    for (int k0 = 0; k0 < K; k0 += 32) {
        __syncthreads();            // previous iteration's fragment reads done
        async_load_b128(ldsA, A + (rowBase + r) * (size_t)K + k0 + cc);
        async_load_b128(ldsB, W + (colBase + r) * (size_t)K + k0 + cc);
        if (k0 + 32 < K) {          // global_prefetch of next K tiles
            __builtin_prefetch(A + (rowBase + r) * (size_t)K + k0 + 32 + cc, 0, 1);
            __builtin_prefetch(W + (colBase + r) * (size_t)K + k0 + 32 + cc, 0, 1);
        }
        wait_async0();
        __syncthreads();

        // A fragment: lane holds row wr*16+l16; K = kb..kb+15 contiguous
        BF16Frag a;
        {
            const unsigned short* ar = &As[wr * 16 + l16][0];
            const int kb = half * 16;
            #pragma unroll
            for (int j = 0; j < 8; ++j) { a.u[j] = ar[kb + j]; a.u[8 + j] = ar[kb + 8 + j]; }
        }
        // B fragments: lane holds col n = l16 (+16); K = kb2..kb2+15 contiguous
        BF16Frag b0, b1;
        {
            const int kb2 = half * 16;
            const unsigned short* br0 = &Bs[wc * 32 + l16][0];
            const unsigned short* br1 = &Bs[wc * 32 + 16 + l16][0];
            #pragma unroll
            for (int j = 0; j < 16; ++j) { b0.u[j] = br0[kb2 + j]; b1.u[j] = br1[kb2 + j]; }
        }
        c0.v = wmma_bf16(a, b0, c0.v);
        c1.v = wmma_bf16(a, b1, c1.v);
    }

    // C layout: vgpr v, lane(l16,half) -> m = v + half*8, n = l16
    const size_t m0 = rowBase + wr * 16 + half * 8;
    const size_t n0 = colBase + wc * 32 + l16;
    #pragma unroll
    for (int v = 0; v < 8; ++v) {
        C[(m0 + v) * (size_t)N + n0]      = c0.e[v];
        C[(m0 + v) * (size_t)N + n0 + 16] = c1.e[v];
    }
}

// ---------------------------------------------------------------------------
// Per-head RMSNorm + RoPE, fp32 [B*S, NH*128] -> bf16 [B, NH, S, 128].
// One wave32 per (b,s,h); lane holds 4 contiguous d values.
// ---------------------------------------------------------------------------
__global__ __launch_bounds__(128)
void norm_rope_kernel(const float* __restrict__ in,
                      const float* __restrict__ cosT, const float* __restrict__ sinT,
                      const float* __restrict__ w,
                      unsigned short* __restrict__ out, int S, int NH) {
    const int wave = blockIdx.x * 4 + (threadIdx.x >> 5);
    const int lane = threadIdx.x & 31;
    const int h  = wave % NH;
    const int bs = wave / NH;       // 0..B*S-1
    const int s  = bs % S;
    const int d0 = lane * 4;

    const float4 x4 = *(const float4*)(in + (size_t)bs * NH * 128 + h * 128 + d0);
    float t[4] = {x4.x, x4.y, x4.z, x4.w};
    float ss = t[0]*t[0] + t[1]*t[1] + t[2]*t[2] + t[3]*t[3];
    #pragma unroll
    for (int o = 1; o < 32; o <<= 1) ss += __shfl_xor(ss, o, 32);
    const float r = rsqrtf(ss * (1.0f / 128.0f) + 1e-6f);

    const float4 c4 = *(const float4*)(cosT + (size_t)s * 128 + d0);
    const float4 s4 = *(const float4*)(sinT + (size_t)s * 128 + d0);
    const float4 w4 = *(const float4*)(w + d0);
    float tn[4] = { t[0]*r*w4.x, t[1]*r*w4.y, t[2]*r*w4.z, t[3]*r*w4.w };
    const float cc[4] = {c4.x, c4.y, c4.z, c4.w};
    const float sn[4] = {s4.x, s4.y, s4.z, s4.w};

    unsigned short o4[4];
    #pragma unroll
    for (int j = 0; j < 4; ++j) {
        // rotate_half: d<64 -> -t[d+64], else t[d-64]; partner lane = lane^16
        const float partner = __shfl_xor(tn[j], 16, 32);
        const float rot = (lane < 16) ? -partner : partner;
        o4[j] = f2bf(tn[j] * cc[j] + rot * sn[j]);
    }
    unsigned short* d = out + (((size_t)(bs / S) * NH + h) * (size_t)S + s) * 128 + d0;
    *(us4*)d = (us4){o4[0], o4[1], o4[2], o4[3]};
}

// fp32 [B,S,NH,128] -> bf16 [B,NH,S,128]
__global__ __launch_bounds__(256)
void v_cast_kernel(const float* __restrict__ in, unsigned short* __restrict__ out,
                   int S, int NH) {
    const size_t idx = (size_t)blockIdx.x * 256 + threadIdx.x;  // B*S*NH*32 threads
    const int d0 = (int)(idx & 31) * 4;
    const size_t v = idx >> 5;
    const int h  = (int)(v % NH);
    const size_t bs = v / NH;
    const int s  = (int)(bs % S);
    const int b  = (int)(bs / S);
    const float4 x4 = *(const float4*)(in + (bs * NH + h) * 128 + d0);
    unsigned short* dst = out + (((size_t)b * NH + h) * (size_t)S + s) * 128 + d0;
    *(us4*)dst = (us4){f2bf(x4.x), f2bf(x4.y), f2bf(x4.z), f2bf(x4.w)};
}

// ---------------------------------------------------------------------------
// Causal flash attention.  One wave32 per (b, h, 16-row q tile).
// Q[B,H,S,128] bf16, K/V[B,KV,S,128] bf16, O bf16 [B,S,H*128].
// Scores QK^T: 8 WMMAs / 32-wide kv block; online softmax; PV: 8 WMMAs.
// V blocks staged row-major in LDS via async DMA overlapped with the score
// WMMAs; P re-laid out through LDS into an A fragment.
// ---------------------------------------------------------------------------
__global__ __launch_bounds__(32)
void flash_attn_wmma(const unsigned short* __restrict__ Q,
                     const unsigned short* __restrict__ Kd,
                     const unsigned short* __restrict__ Vd,
                     unsigned short* __restrict__ O, int S) {
    __shared__ unsigned short Ptile[16][32];   // P row-major for A-frag reload
    __shared__ unsigned short Vs[32][128];     // V block row-major [kv][d]

    const int lane = threadIdx.x;
    const int half = lane >> 4;
    const int l16  = lane & 15;
    const int qt = blockIdx.x;     // 0..S/16-1
    const int h  = blockIdx.y;     // 0..15
    const int b  = blockIdx.z;     // 0..3
    const int hk = h >> 1;         // H/KV == 2
    const int qbase = qt * 16;

    const unsigned short* qp = Q  + (((size_t)b * 16 + h) * S + qbase) * 128;
    const unsigned short* kp = Kd + (((size_t)b * 8 + hk) * S) * 128;
    const unsigned short* vp = Vd + (((size_t)b * 8 + hk) * S) * 128;

    // Resident Q fragments: 4 K-chunks of 32 over D=128
    BF16Frag qf[4];
    #pragma unroll
    for (int kc = 0; kc < 4; ++kc) {
        const unsigned short* row = qp + (size_t)l16 * 128 + kc * 32;
        const int kb = half * 16;
        #pragma unroll
        for (int j = 0; j < 8; ++j) { qf[kc].u[j] = row[kb + j]; qf[kc].u[8 + j] = row[kb + 8 + j]; }
    }

    F32Frag o[8];
    #pragma unroll
    for (int f = 0; f < 8; ++f)
        #pragma unroll
        for (int v = 0; v < 8; ++v) o[f].e[v] = 0.f;
    float m_i[8], l_i[8];
    #pragma unroll
    for (int v = 0; v < 8; ++v) { m_i[v] = -1e30f; l_i[v] = 0.f; }

    const float scale = 0.08838834764831845f;   // 1/sqrt(128)
    const int qmax = qbase + 15;

    for (int sk = 0; sk <= qmax; sk += 32) {
        // ---- async-stage the 32x128 V block (overlaps with score WMMAs) ----
        #pragma unroll
        for (int i = 0; i < 8; ++i) {
            const int idx = i * 32 + lane;      // 256 x 16B chunks
            const int row = idx >> 3;           // 0..31
            const int col = (idx & 7) * 16;     // 0..112
            async_load_b128(lds_off(&Vs[row][col]),
                            vp + (size_t)(sk + row) * 128 + col);
        }

        // ---- scores S = Q @ K_blk^T (16 x 32) ----
        F32Frag s0, s1;
        #pragma unroll
        for (int v = 0; v < 8; ++v) { s0.e[v] = 0.f; s1.e[v] = 0.f; }
        #pragma unroll
        for (int kc = 0; kc < 4; ++kc) {
            BF16Frag kb0, kb1;
            const int kb2 = half * 16;
            const unsigned short* kr0 = kp + (size_t)(sk + l16) * 128 + kc * 32 + kb2;
            const unsigned short* kr1 = kp + (size_t)(sk + 16 + l16) * 128 + kc * 32 + kb2;
            #pragma unroll
            for (int j = 0; j < 16; ++j) { kb0.u[j] = kr0[j]; kb1.u[j] = kr1[j]; }
            s0.v = wmma_bf16(qf[kc], kb0, s0.v);
            s1.v = wmma_bf16(qf[kc], kb1, s1.v);
        }

        // ---- causal mask, row max (reduce over 16 lanes = 16 cols) ----
        float p0[8], p1[8], rowmax[8];
        #pragma unroll
        for (int v = 0; v < 8; ++v) {
            const int sq = qbase + v + half * 8;
            float a0 = (sk + l16      <= sq) ? s0.e[v] * scale : -__builtin_inff();
            float a1 = (sk + 16 + l16 <= sq) ? s1.e[v] * scale : -__builtin_inff();
            p0[v] = a0; p1[v] = a1;
            float mx = fmaxf(a0, a1);
            #pragma unroll
            for (int ofs = 1; ofs < 16; ofs <<= 1) mx = fmaxf(mx, __shfl_xor(mx, ofs, 32));
            rowmax[v] = mx;
        }

        // ---- online softmax update ----
        float alpha[8];
        #pragma unroll
        for (int v = 0; v < 8; ++v) {
            const float mn = fmaxf(m_i[v], rowmax[v]);
            alpha[v] = __expf(m_i[v] - mn);
            m_i[v] = mn;
            p0[v] = __expf(p0[v] - mn);
            p1[v] = __expf(p1[v] - mn);
            float rs = p0[v] + p1[v];
            #pragma unroll
            for (int ofs = 1; ofs < 16; ofs <<= 1) rs += __shfl_xor(rs, ofs, 32);
            l_i[v] = l_i[v] * alpha[v] + rs;
        }
        #pragma unroll
        for (int f = 0; f < 8; ++f)
            #pragma unroll
            for (int v = 0; v < 8; ++v) o[f].e[v] *= alpha[v];

        // ---- stage P (bf16) in LDS; wait for V DMA ----
        #pragma unroll
        for (int v = 0; v < 8; ++v) {
            Ptile[v + half * 8][l16]      = f2bf(p0[v]);
            Ptile[v + half * 8][16 + l16] = f2bf(p1[v]);
        }
        wait_async0();
        __syncthreads();

        // ---- O += P(16x32) @ V(32x128) ----
        BF16Frag pa;
        {
            const unsigned short* pr = &Ptile[l16][0];
            const int kb = half * 16;
            #pragma unroll
            for (int j = 0; j < 8; ++j) { pa.u[j] = pr[kb + j]; pa.u[8 + j] = pr[kb + 8 + j]; }
        }
        #pragma unroll
        for (int f = 0; f < 8; ++f) {
            BF16Frag bv;
            const int kb2 = half * 16;
            #pragma unroll
            for (int j = 0; j < 16; ++j) bv.u[j] = Vs[kb2 + j][f * 16 + l16];
            o[f].v = wmma_bf16(pa, bv, o[f].v);
        }
        __syncthreads();
    }

    // ---- epilogue: (O / l_i) -> bf16 [B,S,H*128] ----
    #pragma unroll
    for (int v = 0; v < 8; ++v) {
        const float inv = 1.0f / l_i[v];
        const int srow = qbase + v + half * 8;
        unsigned short* orow = O + ((size_t)b * S + srow) * 2048 + h * 128;
        #pragma unroll
        for (int f = 0; f < 8; ++f) orow[f * 16 + l16] = f2bf(o[f].e[v] * inv);
    }
}

// ---------------------------------------------------------------------------
extern "C" void kernel_launch(void* const* d_in, const int* in_sizes, int n_in,
                              void* d_out, int out_size, void* d_ws, size_t ws_size,
                              hipStream_t stream) {
    const float* x    = (const float*)d_in[0];
    const float* cosT = (const float*)d_in[1];
    const float* sinT = (const float*)d_in[2];
    const float* wq   = (const float*)d_in[3];
    const float* wk   = (const float*)d_in[4];
    const float* wv   = (const float*)d_in[5];
    const float* wo   = (const float*)d_in[6];
    const float* qw   = (const float*)d_in[7];
    const float* kw   = (const float*)d_in[8];
    float* out = (float*)d_out;

    const int B = 4, S = 2048, HS = 2048, H = 16, KV = 8;
    const int M = B * S;   // 8192

    char* ws = (char*)d_ws;
    size_t off = 0;
    unsigned short* x_b  = (unsigned short*)(ws + off); off += (size_t)M * 2048 * 2;
    unsigned short* wq_b = (unsigned short*)(ws + off); off += (size_t)2048 * 2048 * 2;
    unsigned short* wk_b = (unsigned short*)(ws + off); off += (size_t)1024 * 2048 * 2;
    unsigned short* wv_b = (unsigned short*)(ws + off); off += (size_t)1024 * 2048 * 2;
    unsigned short* wo_b = (unsigned short*)(ws + off); off += (size_t)2048 * 2048 * 2;
    float* q_raw = (float*)(ws + off); off += (size_t)M * 2048 * sizeof(float);
    float* k_raw = (float*)(ws + off); off += (size_t)M * 1024 * sizeof(float);
    float* v_raw = (float*)(ws + off); off += (size_t)M * 1024 * sizeof(float);
    unsigned short* q_b = (unsigned short*)(ws + off); off += (size_t)M * 2048 * 2;
    unsigned short* k_b = (unsigned short*)(ws + off); off += (size_t)M * 1024 * 2;
    unsigned short* v_b = (unsigned short*)(ws + off); off += (size_t)M * 1024 * 2;
    unsigned short* attn_b = (unsigned short*)(ws + off); off += (size_t)M * 2048 * 2;
    (void)ws_size; (void)in_sizes; (void)n_in; (void)out_size;

    // One-shot bf16 precasts (bandwidth-bound; frees the GEMM loops of cvt)
    cast_f32_bf16<<<dim3((unsigned)((size_t)M * 2048 / 1024)), 256, 0, stream>>>(x,  x_b);
    cast_f32_bf16<<<dim3(2048 * 2048 / 1024), 256, 0, stream>>>(wq, wq_b);
    cast_f32_bf16<<<dim3(1024 * 2048 / 1024), 256, 0, stream>>>(wk, wk_b);
    cast_f32_bf16<<<dim3(1024 * 2048 / 1024), 256, 0, stream>>>(wv, wv_b);
    cast_f32_bf16<<<dim3(2048 * 2048 / 1024), 256, 0, stream>>>(wo, wo_b);

    // QKV projections (bf16 WMMA, f32 accumulate)
    gemm_bf16_wmma<<<dim3(M / 64, 2048 / 64), 256, 0, stream>>>(x_b, wq_b, q_raw, M, 2048, HS);
    gemm_bf16_wmma<<<dim3(M / 64, 1024 / 64), 256, 0, stream>>>(x_b, wk_b, k_raw, M, 1024, HS);
    gemm_bf16_wmma<<<dim3(M / 64, 1024 / 64), 256, 0, stream>>>(x_b, wv_b, v_raw, M, 1024, HS);

    // RMSNorm + RoPE + layout transform to [B, heads, S, D] bf16
    norm_rope_kernel<<<dim3(M * H  / 4), 128, 0, stream>>>(q_raw, cosT, sinT, qw, q_b, S, H);
    norm_rope_kernel<<<dim3(M * KV / 4), 128, 0, stream>>>(k_raw, cosT, sinT, kw, k_b, S, KV);
    v_cast_kernel  <<<dim3(M * KV * 32 / 256), 256, 0, stream>>>(v_raw, v_b, S, KV);

    // Causal flash attention (WMMA for QK^T and PV; async V staging)
    flash_attn_wmma<<<dim3(S / 16, H, B), 32, 0, stream>>>(q_b, k_b, v_b, attn_b, S);

    // Output projection (bf16 A from attention, bf16 W, f32 out)
    gemm_bf16_wmma<<<dim3(M / 64, 2048 / 64), 256, 0, stream>>>(attn_b, wo_b, out, M, 2048, HS);
}
